// Seq2Seq_74947179315611
// MI455X (gfx1250) — compile-verified, matrix operand order checked by reading, not run on hardware
//
#include <hip/hip_runtime.h>

// ---------------------------------------------------------------------------
// Seq2Seq LSTM (enc 2xLSTM -> dec 2xLSTM + FC) for MI455X / gfx1250.
// Persistent-workgroup design: 8 WGs x 512 threads (16 wave32 waves).
// Each WG owns a 16-row batch tile for all 512 timesteps.
// Matmuls: v_wmma_f32_16x16x32_bf16, weights pre-converted to bf16 in d_ws.
// ---------------------------------------------------------------------------

constexpr int kH   = 512;    // hidden
constexpr int kG   = 2048;   // 4*H gate width
constexpr int kB   = 128;    // batch
constexpr int kT   = 256;    // encoder seq len
constexpr int kIn  = 6;      // in/out feature dim
constexpr int kOut = 256;    // decoder steps
constexpr int kWN  = kG * kH; // elems per big weight matrix (2^20)

typedef __attribute__((ext_vector_type(16))) __bf16    v16bf;
typedef __attribute__((ext_vector_type(8)))  float     v8f;
typedef __attribute__((ext_vector_type(4)))  unsigned  u32x4;

union BfVec { u32x4 u[2]; v16bf v; };

__device__ __forceinline__ unsigned short f2bf(float f) {
  union { float f; unsigned u; } x; x.f = f;
  unsigned r = x.u + 0x7FFFu + ((x.u >> 16) & 1u);   // round-to-nearest-even
  return (unsigned short)(r >> 16);
}
__device__ __forceinline__ float bf2f(unsigned short h) {
  union { unsigned u; float f; } x; x.u = ((unsigned)h) << 16;
  return x.f;
}
__device__ __forceinline__ float sigmf(float x) { return 1.0f / (1.0f + __expf(-x)); }

// --- prep: convert 6 big f32 weight matrices (2048x512 each) to bf16 in ws ---
__global__ void wcvt_kernel(const float* __restrict__ s0, const float* __restrict__ s1,
                            const float* __restrict__ s2, const float* __restrict__ s3,
                            const float* __restrict__ s4, const float* __restrict__ s5,
                            unsigned short* __restrict__ dst) {
  long long idx = (long long)blockIdx.x * blockDim.x + threadIdx.x;
  if (idx >= 6LL * kWN) return;
  int which = (int)(idx >> 20);
  int off   = (int)(idx & (kWN - 1));
  const float* s = which == 0 ? s0 : which == 1 ? s1 : which == 2 ? s2
                 : which == 3 ? s3 : which == 4 ? s4 : s5;
  dst[idx] = f2bf(s[off]);
}

// ---------------------------------------------------------------------------
// acc[t] (t = gate*2 + s) accumulates a 16x16 f32 tile of the gate matrix.
// A (16x32 bf16) comes from LDS h-buffer; B (32x16 bf16) from bf16 weights,
// stored row-major (N=2048 rows, K=512 cols) so gate[m][n] = sum_k h[m][k]*W[n][k].
// Per-lane layout (ISA 7.12.2, 16-bit A/B): lane<16 -> K{0..7,16..23},
// lane>=16 -> K{8..15,24..31}; two 16-byte loads each.
// ---------------------------------------------------------------------------
__device__ __forceinline__ void mm512(v8f acc[8],
                                      const unsigned short* __restrict__ W,
                                      const unsigned short* hsrc,
                                      const int nrow[8], int l16, int sel) {
  const unsigned short* arow = hsrc + l16 * kH;
  for (int kt = 0; kt < 16; ++kt) {
    const int kb = kt * 32 + sel;
    BfVec a;
    a.u[0] = *(const u32x4*)(arow + kb);
    a.u[1] = *(const u32x4*)(arow + kb + 16);
#pragma unroll
    for (int t = 0; t < 8; ++t) {
      const unsigned short* wp = W + (long long)nrow[t] * kH + kb;
      BfVec b;
      b.u[0] = *(const u32x4*)(wp);
      b.u[1] = *(const u32x4*)(wp + 16);
      acc[t] = __builtin_amdgcn_wmma_f32_16x16x32_bf16(
          false, a.v, false, b.v, (short)0, acc[t], false, false);
    }
  }
}

__device__ __forceinline__ void zero_acc(v8f acc[8]) {
#pragma unroll
  for (int t = 0; t < 8; ++t)
#pragma unroll
    for (int r = 0; r < 8; ++r) acc[t][r] = 0.0f;
}

// Fuse: + bias (+ optional K=6 input projection), activations, c/h update,
// write h back to LDS as bf16. c-state slice stays in VGPRs across all steps.
__device__ __forceinline__ void lstm_gates(v8f acc[8], v8f c[2],
    const float* __restrict__ bih, const float* __restrict__ bhh,
    const float* __restrict__ Wih,      // 2048x6 f32, or nullptr
    const float* xrow,                  // LDS 16x6 f32, or nullptr
    unsigned short* hdst, int col_base, int l16, int half) {
#pragma unroll
  for (int s = 0; s < 2; ++s) {
    const int col = col_base + s * 16 + l16;
    float bI = bih[col]          + bhh[col];
    float bF = bih[kH + col]     + bhh[kH + col];
    float bG = bih[2 * kH + col] + bhh[2 * kH + col];
    float bO = bih[3 * kH + col] + bhh[3 * kH + col];
    float wI[kIn], wF[kIn], wG[kIn], wO[kIn];
    if (Wih) {
#pragma unroll
      for (int k = 0; k < kIn; ++k) {
        wI[k] = Wih[(0 * kH + col) * kIn + k];
        wF[k] = Wih[(1 * kH + col) * kIn + k];
        wG[k] = Wih[(2 * kH + col) * kIn + k];
        wO[k] = Wih[(3 * kH + col) * kIn + k];
      }
    }
    v8f cc = c[s];
#pragma unroll
    for (int r = 0; r < 8; ++r) {
      const int m = r + 8 * half;
      float gi = acc[0 + s][r] + bI;
      float gf = acc[2 + s][r] + bF;
      float gg = acc[4 + s][r] + bG;
      float go = acc[6 + s][r] + bO;
      if (Wih) {
#pragma unroll
        for (int k = 0; k < kIn; ++k) {
          const float xv = xrow[m * kIn + k];
          gi += xv * wI[k]; gf += xv * wF[k];
          gg += xv * wG[k]; go += xv * wO[k];
        }
      }
      const float I  = sigmf(gi);
      const float F  = sigmf(gf);
      const float Gt = tanhf(gg);
      const float O  = sigmf(go);
      const float c2 = F * cc[r] + I * Gt;
      cc[r] = c2;
      hdst[m * kH + col] = f2bf(O * tanhf(c2));
    }
    c[s] = cc;
  }
}

__global__ __launch_bounds__(512, 1)
void seq2seq_kernel(const float* __restrict__ src,
    const float* __restrict__ encWih0,
    const float* __restrict__ enc_bih0, const float* __restrict__ enc_bhh0,
    const float* __restrict__ enc_bih1, const float* __restrict__ enc_bhh1,
    const float* __restrict__ decWih0,
    const float* __restrict__ dec_bih0, const float* __restrict__ dec_bhh0,
    const float* __restrict__ dec_bih1, const float* __restrict__ dec_bhh1,
    const float* __restrict__ fcW, const float* __restrict__ fcb,
    const unsigned short* __restrict__ wEncWhh0,
    const unsigned short* __restrict__ wEncWih1,
    const unsigned short* __restrict__ wEncWhh1,
    const unsigned short* __restrict__ wDecWhh0,
    const unsigned short* __restrict__ wDecWih1,
    const unsigned short* __restrict__ wDecWhh1,
    float* __restrict__ out) {
  __shared__ unsigned short h0buf[16 * kH];  // 16 KB bf16
  __shared__ unsigned short h1buf[16 * kH];  // 16 KB bf16
  __shared__ float          xbuf[16 * kIn];  // current step input (16x6)

  const int tid      = threadIdx.x;
  const int wave     = tid >> 5;
  const int lane     = tid & 31;
  const int l16      = lane & 15;
  const int half     = lane >> 4;
  const int sel      = half * 8;        // K-chunk select per ISA A/B layout
  const int col_base = wave * 32;       // 32 h-columns owned by this wave
  const int m0       = blockIdx.x * 16; // batch-tile base row

  int nrow[8];
#pragma unroll
  for (int t = 0; t < 8; ++t)
    nrow[t] = (t >> 1) * kH + col_base + (t & 1) * 16 + l16;

  for (int i = tid; i < 16 * kH; i += blockDim.x) { h0buf[i] = 0; h1buf[i] = 0; }

  v8f c0[2], c1[2];
#pragma unroll
  for (int s = 0; s < 2; ++s)
#pragma unroll
    for (int r = 0; r < 8; ++r) { c0[s][r] = 0.0f; c1[s][r] = 0.0f; }

  __syncthreads();

  // ------------------------------ encoder ------------------------------
  for (int t = 0; t < kT; ++t) {
    if (tid < 16 * kIn) {
      const int m = tid / kIn, k = tid % kIn;
      xbuf[tid] = src[((long long)(m0 + m) * kT + t) * kIn + k];
    }
    __syncthreads();

    v8f acc[8]; zero_acc(acc);
    mm512(acc, wEncWhh0, h0buf, nrow, l16, sel);
    __syncthreads();                                    // done reading old h0
    lstm_gates(acc, c0, enc_bih0, enc_bhh0, encWih0, xbuf, h0buf,
               col_base, l16, half);
    __syncthreads();                                    // new h0 visible

    zero_acc(acc);
    mm512(acc, wEncWih1, h0buf, nrow, l16, sel);        // x-part (K=512)
    mm512(acc, wEncWhh1, h1buf, nrow, l16, sel);        // h-part (K=512)
    __syncthreads();                                    // done reading old h1
    lstm_gates(acc, c1, enc_bih1, enc_bhh1, nullptr, nullptr, h1buf,
               col_base, l16, half);
    __syncthreads();                                    // new h1 visible
  }

  // decoder seed: last encoder-input timestep
  if (tid < 16 * kIn) {
    const int m = tid / kIn, k = tid % kIn;
    xbuf[tid] = src[((long long)(m0 + m) * kT + (kT - 1)) * kIn + k];
  }
  __syncthreads();

  // ------------------------------ decoder ------------------------------
  for (int td = 0; td < kOut; ++td) {
    v8f acc[8]; zero_acc(acc);
    mm512(acc, wDecWhh0, h0buf, nrow, l16, sel);
    __syncthreads();
    lstm_gates(acc, c0, dec_bih0, dec_bhh0, decWih0, xbuf, h0buf,
               col_base, l16, half);
    __syncthreads();

    zero_acc(acc);
    mm512(acc, wDecWih1, h0buf, nrow, l16, sel);
    mm512(acc, wDecWhh1, h1buf, nrow, l16, sel);
    __syncthreads();
    lstm_gates(acc, c1, dec_bih1, dec_bhh1, nullptr, nullptr, h1buf,
               col_base, l16, half);
    __syncthreads();

    // FC head: 16 rows x 6 outputs, K=512; writes output + feedback input
    if (tid < 16 * kIn) {
      const int m = tid / kIn, j = tid % kIn;
      float sum = fcb[j];
      const float* wr = fcW + j * kH;
#pragma unroll 8
      for (int k = 0; k < kH; ++k) sum += bf2f(h1buf[m * kH + k]) * wr[k];
      out[((long long)(m0 + m) * kOut + td) * kIn + j] = sum;
      xbuf[tid] = sum;
    }
    __syncthreads();
  }
}

extern "C" void kernel_launch(void* const* d_in, const int* in_sizes, int n_in,
                              void* d_out, int out_size, void* d_ws, size_t ws_size,
                              hipStream_t stream) {
  (void)in_sizes; (void)n_in; (void)out_size; (void)ws_size;
  const float* src      = (const float*)d_in[0];
  const float* encWih0  = (const float*)d_in[1];
  const float* encWhh0  = (const float*)d_in[2];
  const float* enc_bih0 = (const float*)d_in[3];
  const float* enc_bhh0 = (const float*)d_in[4];
  const float* encWih1  = (const float*)d_in[5];
  const float* encWhh1  = (const float*)d_in[6];
  const float* enc_bih1 = (const float*)d_in[7];
  const float* enc_bhh1 = (const float*)d_in[8];
  const float* decWih0  = (const float*)d_in[9];
  const float* decWhh0  = (const float*)d_in[10];
  const float* dec_bih0 = (const float*)d_in[11];
  const float* dec_bhh0 = (const float*)d_in[12];
  const float* decWih1  = (const float*)d_in[13];
  const float* decWhh1  = (const float*)d_in[14];
  const float* dec_bih1 = (const float*)d_in[15];
  const float* dec_bhh1 = (const float*)d_in[16];
  const float* fcW      = (const float*)d_in[17];
  const float* fcb      = (const float*)d_in[18];

  unsigned short* wbf = (unsigned short*)d_ws;   // 6 * 2^20 bf16 = 12 MB
  const int total = 6 * kWN;
  wcvt_kernel<<<(total + 255) / 256, 256, 0, stream>>>(
      encWhh0, encWih1, encWhh1, decWhh0, decWih1, decWhh1, wbf);

  seq2seq_kernel<<<dim3(kB / 16), dim3(512), 0, stream>>>(
      src, encWih0, enc_bih0, enc_bhh0, enc_bih1, enc_bhh1,
      decWih0, dec_bih0, dec_bhh0, dec_bih1, dec_bhh1, fcW, fcb,
      wbf + 0 * (size_t)kWN, wbf + 1 * (size_t)kWN, wbf + 2 * (size_t)kWN,
      wbf + 3 * (size_t)kWN, wbf + 4 * (size_t)kWN, wbf + 5 * (size_t)kWN,
      (float*)d_out);
}
